// OptimizedMultiMetaFingerNet_40785009443179
// MI455X (gfx1250) — matrix operand order repaired and checked by reading, not compiled, not run on hardware
//
#include <hip/hip_runtime.h>
#include <hip/hip_bf16.h>
#include <stdint.h>

// MI455X / gfx1250, wave32. One workgroup (256 thr = 8 waves) per batch item.
// Whole 128x256 activation + 128x768 QKV kept resident in LDS (290KB of 320KB).
// All GEMMs via v_wmma_f32_16x16x32_bf16 (f32 accumulate).
// All global reads through addrspace(1) (global_load_*, LOADcnt-only path).

typedef __attribute__((ext_vector_type(16))) __bf16 v16bf;
typedef __attribute__((ext_vector_type(8)))  float  v8f;
typedef __attribute__((ext_vector_type(4)))  unsigned int u32x4;   // plain vector (not HIP class)

#define LDX    264      // padded row stride for xb (bf16 elems), 16B-aligned rows
#define NSEQ   128
#define CDIM   256
#define QKVD   768
#define HDIM   32
#define TOPM   20

static __device__ __forceinline__ v8f wmma_bf16(v16bf a, v16bf b, v8f c) {
  return __builtin_amdgcn_wmma_f32_16x16x32_bf16(false, a, false, b, (short)0, c, false, false);
}

// Explicit global-address-space loads: force global_load_* (LOADcnt-only),
// instead of flat_load_* which also ties up DScnt / the LDS datapath.
typedef __attribute__((address_space(1))) const u32x4 gu32x4;
typedef __attribute__((address_space(1))) const float gfloat;
static __device__ __forceinline__ u32x4 gload16B(const void* p) {
  return *(const gu32x4*)(unsigned long long)(uintptr_t)p;
}
static __device__ __forceinline__ float gloadf(const float* p) {
  return *(const gfloat*)(unsigned long long)(uintptr_t)p;
}

// bf16 pair <-> float helpers (bf16 -> f32 is exactly u16 << 16)
static __device__ __forceinline__ float bf_lo(unsigned int pk) {
  return __builtin_bit_cast(float, pk << 16);
}
static __device__ __forceinline__ float bf_hi(unsigned int pk) {
  return __builtin_bit_cast(float, pk & 0xffff0000u);
}
static __device__ __forceinline__ unsigned int pack_bf16(float a, float b) {
  union { __bf16 e[2]; unsigned int u; } z;
  z.e[0] = (__bf16)a; z.e[1] = (__bf16)b;
  return z.u;
}

// A fragment (16 x 32 bf16) from row-major LDS buffer, row stride ld.
// Layout: lane m=L%16, half=L/16; dword j -> K=2j+16*half (matches B fragments).
// => two contiguous 16B reads at col k0+16*half and k0+16*half+8.
static __device__ __forceinline__ v16bf load_a_frag(const __bf16* buf, int row0, int k0, int ld, int lane) {
  int m = lane & 15, half = lane >> 4;
  const __bf16* p = buf + (size_t)(row0 + m) * ld + k0 + 16 * half;
  union { u32x4 q[2]; v16bf v; } u;
  u.q[0] = *(const u32x4*)p;
  u.q[1] = *(const u32x4*)(p + 8);
  return u.v;
}

// B fragment from pre-swizzled global weight tile: 32 lanes x 16 contiguous bf16.
static __device__ __forceinline__ v16bf load_b_packed(const __bf16* tile, int lane) {
  const __bf16* p = tile + lane * 16;
  union { u32x4 q[2]; v16bf v; } u;
  u.q[0] = gload16B(p);
  u.q[1] = gload16B(p + 8);
  return u.v;
}

// B fragment (32 x 16) where B[k][n] = buf[(n0+n)*ld + c0 + k]  (K^T from row-major K in LDS).
// B layout used here (same as A): lane n=L%16, half=L/16, dword j -> K = 2j + 16*half  => 16 contiguous bf16.
static __device__ __forceinline__ v16bf load_b_contig(const __bf16* buf, int n0, int c0, int ld, int lane) {
  int n = lane & 15, half = lane >> 4;
  const __bf16* p = buf + (size_t)(n0 + n) * ld + c0 + 16 * half;
  union { u32x4 q[2]; v16bf v; } u;
  u.q[0] = *(const u32x4*)p;
  u.q[1] = *(const u32x4*)(p + 8);
  return u.v;
}

// B fragment (32 x 16) from row-major V in LDS: B[k][n] = base[(k0+k)*ld + n].
static __device__ __forceinline__ v16bf load_b_strided(const __bf16* base, int k0, int ld, int lane) {
  int n = lane & 15, half = lane >> 4;
  union { __bf16 e[16]; v16bf v; } u;
#pragma unroll
  for (int j = 0; j < 8; ++j) {
    int k = k0 + 2 * j + 16 * half;
    u.e[2 * j]     = base[(size_t)k * ld + n];
    u.e[2 * j + 1] = base[(size_t)(k + 1) * ld + n];
  }
  return u.v;
}

// Store 16x16 f32 D-tile (VGPR r: row r+8*half, col n=lane%16) as bf16, row-major LDS.
static __device__ __forceinline__ void store_d_bf16(__bf16* buf, int row0, int col0, int ld, int lane, v8f acc) {
  int n = lane & 15, half = lane >> 4;
#pragma unroll
  for (int r = 0; r < 8; ++r)
    buf[(size_t)(row0 + r + 8 * half) * ld + col0 + n] = (__bf16)acc[r];
}

// ---- prep: pack f32 weight (K x N, row-major) into B-fragment-swizzled bf16 tiles.
// tile (nt,kt): 512 elems; elem (lane*16 + 2j..) = W[kt*32 + 2j + 16*(lane>>4)][nt*16 + lane%16]
__global__ void pack_weight_bf16(const float* __restrict__ w, __bf16* __restrict__ dst, int K, int N) {
  int total = (K * N) >> 1;   // packed bf16 pairs
  int KT = K >> 5;
  for (int d = blockIdx.x * blockDim.x + threadIdx.x; d < total; d += gridDim.x * blockDim.x) {
    int j    = d & 7;
    int lane = (d >> 3) & 31;
    int tile = d >> 8;
    int kt = tile % KT;
    int nt = tile / KT;
    int n = nt * 16 + (lane & 15);
    int k = kt * 32 + 2 * j + 16 * (lane >> 4);
    dst[2 * d]     = (__bf16)gloadf(w + (size_t)k * N + n);
    dst[2 * d + 1] = (__bf16)gloadf(w + (size_t)(k + 1) * N + n);
  }
}

__global__ void __launch_bounds__(256)
fused_transformer(const float* __restrict__ xin, const float* __restrict__ pos,
                  const __bf16* __restrict__ wqkv0, const float* __restrict__ bqkv0,
                  const __bf16* __restrict__ wproj0, const float* __restrict__ bproj0,
                  const float* __restrict__ g0, const float* __restrict__ be0,
                  const __bf16* __restrict__ wqkv1, const float* __restrict__ bqkv1,
                  const __bf16* __restrict__ wproj1, const float* __restrict__ bproj1,
                  const float* __restrict__ g1, const float* __restrict__ be1,
                  const float* __restrict__ w1, const float* __restrict__ b1c,
                  const float* __restrict__ w2, const float* __restrict__ b2c,
                  float* __restrict__ out)
{
  extern __shared__ char smem[];
  __bf16* xb   = (__bf16*)smem;                                    // 128 x LDX   (66 KB)
  __bf16* qkvb = (__bf16*)(smem + 128 * LDX * 2);                  // 128 x 768   (192 KB)
  __bf16* sbuf = (__bf16*)(smem + 128 * LDX * 2 + 128 * QKVD * 2); // 8 x 16x128  (32 KB)
  float*  fscr = (float*)sbuf;                                     // reused by classifier

  const int tid  = threadIdx.x;
  const int lane = tid & 31;
  const int wid  = tid >> 5;
  const int blk  = blockIdx.x;

  // ---- load input (B, C=256, N=128), transpose to (N, C), add pos_embed, cvt bf16
  for (int i = 0; i < 128; ++i) {
    int flat = i * 256 + tid;          // coalesced over c-major input
    int c = flat >> 7;
    int n = flat & 127;
    xb[(size_t)n * LDX + c] =
        (__bf16)(gloadf(xin + (size_t)blk * 32768 + flat) + gloadf(pos + (size_t)n * CDIM + c));
  }
  __syncthreads();

  const __bf16* wq[2] = {wqkv0, wqkv1};
  const float*  bq[2] = {bqkv0, bqkv1};
  const __bf16* wp[2] = {wproj0, wproj1};
  const float*  bp[2] = {bproj0, bproj1};
  const float*  lg[2] = {g0, g1};
  const float*  lb[2] = {be0, be1};

  for (int layer = 0; layer < 2; ++layer) {
    // ===== QKV GEMM: (128x256) @ (256x768). Wave w owns N-tiles [6w, 6w+6):
    // hoist the 8 B-fragments once per nt (weights read exactly once per block),
    // stream A-fragments from LDS per row-tile. =====
    {
      const __bf16* wbase = wq[layer];
      const float*  bias  = bq[layer];
      for (int ntl = 0; ntl < 6; ++ntl) {
        int nt = 6 * wid + ntl;
        if (ntl + 1 < 6)  // prefetch next nt's tile stream into L2/L0
          __builtin_prefetch(wbase + (size_t)(nt + 1) * 8 * 512, 0, 1);
        v16bf bf[8];
#pragma unroll
        for (int kt = 0; kt < 8; ++kt)
          bf[kt] = load_b_packed(wbase + (size_t)(nt * 8 + kt) * 512, lane);
        float bv = gloadf(bias + nt * 16 + (lane & 15));
        for (int t = 0; t < 8; ++t) {
          v8f acc = {};
#pragma unroll
          for (int kt = 0; kt < 8; ++kt)
            acc = wmma_bf16(load_a_frag(xb, 16 * t, 32 * kt, LDX, lane), bf[kt], acc);
#pragma unroll
          for (int r = 0; r < 8; ++r) acc[r] += bv;
          store_d_bf16(qkvb, 16 * t, 16 * nt, QKVD, lane, acc);
        }
      }
    }
    __syncthreads();

    // ===== attention: wave h = head h. q cols [h*32), k cols [256+h*32), v cols [512+h*32) =====
    {
      const int h = wid;
      __bf16* srow = sbuf + (size_t)wid * 16 * 128;     // per-wave 16x128 score tile
      const float scale = 0.17677669529663687f;         // 1/sqrt(32)
      for (int t = 0; t < 8; ++t) {                     // query tiles (16 queries each)
        v16bf qf = load_a_frag(qkvb, 16 * t, h * HDIM, QKVD, lane);
        for (int nt = 0; nt < 8; ++nt) {
          v16bf kf = load_b_contig(qkvb, 16 * nt, CDIM + h * HDIM, QKVD, lane);
          v8f z = {};
          v8f s = wmma_bf16(qf, kf, z);
          int n = lane & 15, half = lane >> 4;
#pragma unroll
          for (int r = 0; r < 8; ++r)
            srow[(size_t)(r + 8 * half) * 128 + 16 * nt + n] = (__bf16)(s[r] * scale);
        }
        asm volatile("" ::: "memory");
        // top-20 threshold + softmax; lanes 0..15 each own one query row.
        // Scan packed: one ds_load_b32 delivers two bf16 scores (shift-unpack).
        if (lane < 16) {
          unsigned int* rp32 = (unsigned int*)(srow + (size_t)lane * 128);  // 64 dwords
          float kth  = __builtin_inff();
          float rmax = -__builtin_inff();
          for (int it = 0; it < TOPM; ++it) {           // iterative kth-largest extraction
            float m = -__builtin_inff();
            for (int jj = 0; jj < 64; ++jj) {
              unsigned int pk = rp32[jj];
              float v0 = bf_lo(pk), v1 = bf_hi(pk);
              if (v0 < kth && v0 > m) m = v0;
              if (v1 < kth && v1 > m) m = v1;
            }
            if (it == 0) rmax = m;
            kth = m;
          }
          float sum = 0.f;
          for (int jj = 0; jj < 64; ++jj) {
            unsigned int pk = rp32[jj];
            float v0 = bf_lo(pk), v1 = bf_hi(pk);
            float p0 = (v0 >= kth) ? __expf(v0 - rmax) : 0.f;
            float p1 = (v1 >= kth) ? __expf(v1 - rmax) : 0.f;
            sum += p0 + p1;
            rp32[jj] = pack_bf16(p0, p1);
          }
          float inv = 1.f / sum;
          for (int jj = 0; jj < 64; ++jj) {
            unsigned int pk = rp32[jj];
            rp32[jj] = pack_bf16(bf_lo(pk) * inv, bf_hi(pk) * inv);
          }
        }
        asm volatile("" ::: "memory");
        // o(16x32) = P(16x128) @ V(128x32); overwrite already-consumed q cols with o
        v8f o0 = {}, o1 = {};
#pragma unroll
        for (int ks = 0; ks < 4; ++ks) {
          v16bf pf = load_a_frag(srow, 0, 32 * ks, 128, lane);
          o0 = wmma_bf16(pf, load_b_strided(qkvb + 2 * CDIM + h * HDIM,      32 * ks, QKVD, lane), o0);
          o1 = wmma_bf16(pf, load_b_strided(qkvb + 2 * CDIM + h * HDIM + 16, 32 * ks, QKVD, lane), o1);
        }
        store_d_bf16(qkvb, 16 * t, h * HDIM,      QKVD, lane, o0);
        store_d_bf16(qkvb, 16 * t, h * HDIM + 16, QKVD, lane, o1);
      }
    }
    __syncthreads();

    // ===== proj GEMM + bias + residual: wave w owns N-tiles [2w, 2w+2) (cols 32w..32w+31)
    // => reads & writes of xb are column-disjoint across waves. =====
    {
      const __bf16* wbase = wp[layer];
      const float*  bias  = bp[layer];
      int n = lane & 15, half = lane >> 4;
      for (int ntl = 0; ntl < 2; ++ntl) {
        int nt = 2 * wid + ntl;
        if (ntl == 0)
          __builtin_prefetch(wbase + (size_t)(nt + 1) * 8 * 512, 0, 1);
        v16bf bf[8];
#pragma unroll
        for (int kt = 0; kt < 8; ++kt)
          bf[kt] = load_b_packed(wbase + (size_t)(nt * 8 + kt) * 512, lane);
        float bv = gloadf(bias + nt * 16 + n);
        for (int t = 0; t < 8; ++t) {
          v8f acc = {};
#pragma unroll
          for (int kt = 0; kt < 8; ++kt)
            acc = wmma_bf16(load_a_frag(qkvb, 16 * t, 32 * kt, QKVD, lane), bf[kt], acc);
#pragma unroll
          for (int r = 0; r < 8; ++r)
            acc[r] += bv + (float)xb[(size_t)(16 * t + r + 8 * half) * LDX + nt * 16 + n];
          store_d_bf16(xb, 16 * t, 16 * nt, LDX, lane, acc);
        }
      }
    }
    __syncthreads();

    // ===== layernorm per row; 2 threads per row, pair-reduce with shfl_xor.
    // Packed dword LDS access: 64 loads / 64 stores per half-row. =====
    {
      int r = tid >> 1, half = tid & 1;
      unsigned int* xr32 = (unsigned int*)(xb + (size_t)r * LDX + half * 128);  // 64 dwords
      float s = 0.f, ss = 0.f;
      for (int jj = 0; jj < 64; ++jj) {
        unsigned int pk = xr32[jj];
        float v0 = bf_lo(pk), v1 = bf_hi(pk);
        s += v0 + v1; ss += v0 * v0 + v1 * v1;
      }
      s  += __shfl_xor(s, 1);
      ss += __shfl_xor(ss, 1);
      float mean = s * (1.f / 256.f);
      float var  = ss * (1.f / 256.f) - mean * mean;
      float rstd = rsqrtf(var + 1e-5f);
      const float* g = lg[layer];
      const float* b = lb[layer];
      for (int jj = 0; jj < 64; ++jj) {
        int c = half * 128 + 2 * jj;
        unsigned int pk = xr32[jj];
        float y0 = (bf_lo(pk) - mean) * rstd * gloadf(g + c)     + gloadf(b + c);
        float y1 = (bf_hi(pk) - mean) * rstd * gloadf(g + c + 1) + gloadf(b + c + 1);
        xr32[jj] = pack_bf16(y0, y1);
      }
    }
    __syncthreads();
  }

  // ===== mean over sequence: thread c sums its column =====
  {
    float s = 0.f;
    for (int r = 0; r < NSEQ; ++r)
      s += (float)xb[(size_t)r * LDX + tid];
    fscr[tid] = s * (1.f / 128.f);
  }
  __syncthreads();

  // ===== classifier: h = relu(xm @ W1 + b1); logit = h @ w2 + b2 =====
  float* partial = fscr + 256;
  if (tid < 128) {
    float acc = gloadf(b1c + tid);
    for (int c = 0; c < 256; ++c)
      acc += fscr[c] * gloadf(w1 + (size_t)c * 128 + tid);
    float hrelu = fmaxf(acc, 0.f);
    partial[tid] = hrelu * gloadf(w2 + tid);
  }
  __syncthreads();
  if (tid == 0) {
    float o = gloadf(b2c);
    for (int j = 0; j < 128; ++j) o += partial[j];
    out[blk] = o;
  }
}

extern "C" void kernel_launch(void* const* d_in, const int* in_sizes, int n_in,
                              void* d_out, int out_size, void* d_ws, size_t ws_size,
                              hipStream_t stream) {
  (void)in_sizes; (void)n_in; (void)out_size; (void)ws_size;
  const float* xin = (const float*)d_in[0];
  const float* pos = (const float*)d_in[1];
  const float* w1  = (const float*)d_in[2];
  const float* b1  = (const float*)d_in[3];
  const float* w2  = (const float*)d_in[4];
  const float* b2  = (const float*)d_in[5];
  const float* qw0 = (const float*)d_in[6];
  const float* qb0 = (const float*)d_in[7];
  const float* pw0 = (const float*)d_in[8];
  const float* pb0 = (const float*)d_in[9];
  const float* g0  = (const float*)d_in[10];
  const float* be0 = (const float*)d_in[11];
  const float* qw1 = (const float*)d_in[12];
  const float* qb1 = (const float*)d_in[13];
  const float* pw1 = (const float*)d_in[14];
  const float* pb1 = (const float*)d_in[15];
  const float* g1  = (const float*)d_in[16];
  const float* be1 = (const float*)d_in[17];

  __bf16* ws   = (__bf16*)d_ws;           // 1 MB of packed bf16 weights
  __bf16* wq0p = ws;
  __bf16* wp0p = wq0p + 256 * 768;
  __bf16* wq1p = wp0p + 256 * 256;
  __bf16* wp1p = wq1p + 256 * 768;

  pack_weight_bf16<<<384, 256, 0, stream>>>(qw0, wq0p, 256, 768);
  pack_weight_bf16<<<128, 256, 0, stream>>>(pw0, wp0p, 256, 256);
  pack_weight_bf16<<<384, 256, 0, stream>>>(qw1, wq1p, 256, 768);
  pack_weight_bf16<<<128, 256, 0, stream>>>(pw1, wp1p, 256, 256);

  size_t smem = (size_t)128 * LDX * 2      // xb
              + (size_t)128 * QKVD * 2     // qkv
              + (size_t)8 * 16 * 128 * 2;  // scores (290 KB total, < 320 KB/WGP)
  fused_transformer<<<480, 256, smem, stream>>>(xin, pos,
                                                wq0p, qb0, wp0p, pb0, g0, be0,
                                                wq1p, qb1, wp1p, pb1, g1, be1,
                                                w1, b1, w2, b2, (float*)d_out);
}
